// Dynamic_Lightweight_Convolution_80178449482211
// MI455X (gfx1250) — matrix operand order repaired and checked by reading, not compiled, not run on hardware
//
#include <hip/hip_runtime.h>
#include <hip/hip_bf16.h>

typedef __bf16 bf16_t;
typedef __attribute__((ext_vector_type(16))) __bf16 v16bf;
typedef __attribute__((ext_vector_type(8)))  float  v8f;

#define BB   8
#define SS   2048
#define DD   1024
#define KK   7
#define HH   16
#define LL   2042      // S - K + 1
#define NT   128       // ceil(L / 16)
#define WPB  4         // waves per block in gemm_softmax

// ---------------------------------------------------------------------------
// Kernel 0: repack W (D x 112, f32 row-major) into bf16 laid out exactly as the
// B-operand fragment of v_wmma_f32_16x16x32_bf16 wants it per lane:
//   element index e = ((t*32 + kk)*32 + lane)*16 + i
//   lane: n = column h = lane&15 within N-tile t, K-half = lane>>4
//   i<8  -> K = kk*32 + (lane>>4)*8 + i
//   i>=8 -> K = kk*32 + (lane>>4)*8 + 16 + (i-8)
// so each lane's 16 values are 32 contiguous bytes (two b128 loads).
// ---------------------------------------------------------------------------
__global__ void pack_w_kernel(const float* __restrict__ W, bf16_t* __restrict__ Wb) {
    int e = blockIdx.x * 256 + threadIdx.x;          // 0 .. 7*32*32*16-1 = 114687
    if (e >= 7 * 32 * 32 * 16) return;
    int i    = e & 15;
    int lane = (e >> 4) & 31;
    int kk   = (e >> 9) & 31;
    int t    = e >> 14;                              // N-tile == conv tap k
    int kb   = kk * 32 + (lane >> 4) * 8;
    int Kidx = kb + (i < 8 ? i : i + 8);             // i>=8: kb+16+(i-8)
    int j    = t * 16 + (lane & 15);                 // column in 0..111
    Wb[e] = (bf16_t)W[(size_t)Kidx * (KK * HH) + j];
}

// ---------------------------------------------------------------------------
// Kernel 1: fused projection GEMM (bf16 WMMA, f32 accum) + softmax over taps.
// One wave = one 16-row M tile of (b, l). acc[t] = 16x16 logits for tap t.
// Softmax over t is per-thread (lane = head, vgpr r = row). Weights (already
// scaled by 1/7 for the mean) go to workspace as [b][l][k][h] f32.
// ---------------------------------------------------------------------------
__global__ __launch_bounds__(WPB * 32)
void gemm_softmax_kernel(const float* __restrict__ x,
                         const bf16_t* __restrict__ Wb,
                         const float* __restrict__ bias,
                         float* __restrict__ kernw) {
    const int lane  = threadIdx.x & 31;
    const int wave  = threadIdx.x >> 5;
    const int gtile = blockIdx.x * WPB + wave;       // 0 .. B*NT-1 (exact grid)
    const int bidx  = gtile >> 7;                    // / NT
    const int ltile = gtile & (NT - 1);
    const int l0    = ltile * 16;

    const int m    = lane & 15;                      // A row within tile
    const int half = lane >> 4;                      // K-half select

    int row_l = l0 + m;
    if (row_l > LL - 1) row_l = LL - 1;              // clamp (stores masked later)
    const float* __restrict__ arow =
        x + ((size_t)bidx * SS + (row_l + KK - 1)) * DD;

    float bb[KK];
#pragma unroll
    for (int t = 0; t < KK; ++t) bb[t] = bias[t * 16 + (lane & 15)];

    v8f acc[KK] = {};

    for (int kk = 0; kk < 32; ++kk) {
        const int kb = kk * 32 + half * 8;

        const float4 f0 = *(const float4*)(arow + kb);
        const float4 f1 = *(const float4*)(arow + kb + 4);
        const float4 f2 = *(const float4*)(arow + kb + 16);
        const float4 f3 = *(const float4*)(arow + kb + 20);

        if (kk + 1 < 32) __builtin_prefetch(arow + kb + 32, 0, 1);  // global_prefetch_b8

        v16bf A;
        A[0]  = (bf16_t)f0.x;  A[1]  = (bf16_t)f0.y;
        A[2]  = (bf16_t)f0.z;  A[3]  = (bf16_t)f0.w;
        A[4]  = (bf16_t)f1.x;  A[5]  = (bf16_t)f1.y;
        A[6]  = (bf16_t)f1.z;  A[7]  = (bf16_t)f1.w;
        A[8]  = (bf16_t)f2.x;  A[9]  = (bf16_t)f2.y;
        A[10] = (bf16_t)f2.z;  A[11] = (bf16_t)f2.w;
        A[12] = (bf16_t)f3.x;  A[13] = (bf16_t)f3.y;
        A[14] = (bf16_t)f3.z;  A[15] = (bf16_t)f3.w;

#pragma unroll
        for (int t = 0; t < KK; ++t) {
            const v16bf Bv =
                *(const v16bf*)(Wb + ((((size_t)t * 32 + kk) * 32 + lane) << 4));
            acc[t] = __builtin_amdgcn_wmma_f32_16x16x32_bf16(
                false, A, false, Bv, (short)0, acc[t], false, false);
        }
    }

    // per-thread softmax over the 7 taps, folded 1/7 mean, masked store
#pragma unroll
    for (int r = 0; r < 8; ++r) {
        float lg[KK];
        float mx = -3.0e38f;
#pragma unroll
        for (int t = 0; t < KK; ++t) {
            lg[t] = acc[t][r] + bb[t];
            mx = fmaxf(mx, lg[t]);
        }
        float s = 0.0f;
#pragma unroll
        for (int t = 0; t < KK; ++t) {
            lg[t] = __expf(lg[t] - mx);
            s += lg[t];
        }
        const float inv = 1.0f / (7.0f * s);
        const int l = l0 + r + 8 * half;
        if (l < LL) {
            float* __restrict__ dst =
                kernw + ((size_t)bidx * LL + l) * (KK * HH) + (lane & 15);
#pragma unroll
            for (int t = 0; t < KK; ++t) dst[t * 16] = lg[t] * inv;
        }
    }
}

// ---------------------------------------------------------------------------
// Kernel 2: weighted window sum. Block = (b, 16-l tile, 512-d chunk).
// x halo tile (22 x 512) and the 16x112 weights staged in LDS for 7x reuse.
// ---------------------------------------------------------------------------
__global__ __launch_bounds__(256)
void conv_kernel(const float* __restrict__ x,
                 const float* __restrict__ kernw,
                 float* __restrict__ out) {
    __shared__ __align__(16) float xs[22 * 512];     // 44 KB
    __shared__ __align__(16) float wsm[16 * 112];    // 7 KB

    const int bidx = blockIdx.z;
    const int l0   = blockIdx.y * 16;
    const int d0   = blockIdx.x * 512;

    // stage x tile (rows l0 .. l0+21 clamped to S-1)
    for (int idx = threadIdx.x; idx < 22 * 128; idx += 256) {
        int row = idx >> 7;
        int c   = idx & 127;
        int xr  = l0 + row;
        if (xr > SS - 1) xr = SS - 1;
        *(float4*)&xs[row * 512 + c * 4] =
            *(const float4*)&x[((size_t)bidx * SS + xr) * DD + d0 + c * 4];
    }
    // stage weights
    for (int idx = threadIdx.x; idx < 16 * 112; idx += 256) {
        int r = idx / 112;
        int j = idx - r * 112;
        int l = l0 + r;
        wsm[idx] = (l < LL)
                       ? kernw[((size_t)bidx * LL + l) * 112 + j]
                       : 0.0f;
    }
    __syncthreads();

    const int dq = threadIdx.x & 127;   // float4 index within 512-chunk
    const int rr = threadIdx.x >> 7;    // 0/1: row phase
    const int dc = dq * 4;
    const int h0 = dc & 15;             // d0 is a multiple of 16

    for (int r = rr; r < 16; r += 2) {
        const int l = l0 + r;
        if (l >= LL) break;
        float4 a = make_float4(0.f, 0.f, 0.f, 0.f);
#pragma unroll
        for (int k = 0; k < KK; ++k) {
            const float4 w4 = *(const float4*)&wsm[r * 112 + k * 16 + h0];
            const float4 xv = *(const float4*)&xs[(r + k) * 512 + dc];
            a.x = fmaf(w4.x, xv.x, a.x);
            a.y = fmaf(w4.y, xv.y, a.y);
            a.z = fmaf(w4.z, xv.z, a.z);
            a.w = fmaf(w4.w, xv.w, a.w);
        }
        *(float4*)&out[((size_t)bidx * LL + l) * DD + d0 + dc] = a;
    }
}

// ---------------------------------------------------------------------------
extern "C" void kernel_launch(void* const* d_in, const int* in_sizes, int n_in,
                              void* d_out, int out_size, void* d_ws, size_t ws_size,
                              hipStream_t stream) {
    const float* x    = (const float*)d_in[0];   // (8, 2048, 1024) f32
    const float* W    = (const float*)d_in[1];   // (1024, 112)    f32
    const float* bias = (const float*)d_in[2];   // (112,)         f32
    float* out        = (float*)d_out;           // (8, 2042, 1024) f32

    // workspace: [ Wb bf16 packed : 224 KB ][ kernw f32 : ~7.3 MB ]
    bf16_t* Wb   = (bf16_t*)d_ws;
    float* kernw = (float*)((char*)d_ws + (size_t)(7 * 32 * 32 * 16) * sizeof(bf16_t));

    // 1) pack W into WMMA-B bf16 fragment order
    pack_w_kernel<<<(7 * 32 * 32 * 16 + 255) / 256, 256, 0, stream>>>(W, Wb);

    // 2) fused bf16-WMMA projection + per-thread softmax over taps
    //    grid: B*NT waves / WPB waves-per-block = 8*128/4 = 256 blocks exact
    gemm_softmax_kernel<<<(BB * NT) / WPB, WPB * 32, 0, stream>>>(x, Wb, bias, kernw);

    // 3) LDS-tiled weighted window sum
    conv_kernel<<<dim3(DD / 512, NT, BB), 256, 0, stream>>>(x, kernw, out);
}
